// HeteroGNN_89335319757571
// MI455X (gfx1250) — compile-verified, matrix operand order checked by reading, not run on hardware
//
#include <hip/hip_runtime.h>
#include <hip/hip_bf16.h>

typedef __attribute__((ext_vector_type(16))) __bf16 v16bf;
typedef __attribute__((ext_vector_type(8)))  float  v8f;

#define ND   4096
#define NT   16384
#define BSZ  4096
#define FD   200
#define FDP  224
#define FT   570
#define FTP  576
#define FC   890
#define FCP  896
#define HD   768
#define N_EDD 262144
#define N_EDT 524288
#define N_ETD 524288
#define N_ETT 1048576

__device__ __forceinline__ unsigned short f2bf(float f) {
  union { float f; unsigned u; } v; v.f = f;
  unsigned u = v.u + 0x7FFFu + ((v.u >> 16) & 1u);
  return (unsigned short)(u >> 16);
}
__device__ __forceinline__ float bf2f(unsigned short b) {
  union { unsigned u; float f; } v; v.u = ((unsigned)b) << 16;
  return v.f;
}

// ---------------- input prep: f32 -> bf16 (clean + masked), K padded to Kp ----
__global__ void k_prep(const float* __restrict__ X, const float* __restrict__ mask,
                       const float* __restrict__ token,
                       unsigned short* __restrict__ clean,
                       unsigned short* __restrict__ masked,
                       int rows, int K, int Kp) {
  long long i = (long long)blockIdx.x * blockDim.x + threadIdx.x;
  long long total = (long long)rows * Kp;
  if (i >= total) return;
  int r = (int)(i / Kp), c = (int)(i % Kp);
  float v = (c < K) ? X[(long long)r * K + c] : 0.f;
  clean[i] = f2bf(v);
  float m = mask[r];
  float mv = (c < K) ? (v * (1.f - m) + m * token[c]) : 0.f;
  masked[i] = f2bf(mv);
}

// ---------------- weight convert: W[K x N] f32 -> Bt[N x Kp] bf16 (transposed)
__global__ void k_wconv_t(const float* __restrict__ W, unsigned short* __restrict__ Bt,
                          int K, int N, int Kp) {
  long long i = (long long)blockIdx.x * blockDim.x + threadIdx.x;
  long long total = (long long)N * Kp;
  if (i >= total) return;
  int n = (int)(i / Kp), k = (int)(i % Kp);
  float v = (k < K) ? W[(long long)k * N + n] : 0.f;
  Bt[i] = f2bf(v);
}

// ---------------- wdst_vec[k] = sum_h Wdst[k][h] * adst[h] ---------------------
__global__ void k_colvec(const float* __restrict__ W, const float* __restrict__ a,
                         float* __restrict__ outv, int Hn) {
  __shared__ float sm[256];
  int k = blockIdx.x;
  float s = 0.f;
  for (int h = threadIdx.x; h < Hn; h += 256) s += W[(long long)k * Hn + h] * a[h];
  sm[threadIdx.x] = s; __syncthreads();
  for (int st = 128; st > 0; st >>= 1) {
    if ((int)threadIdx.x < st) sm[threadIdx.x] += sm[threadIdx.x + st];
    __syncthreads();
  }
  if (threadIdx.x == 0) outv[k] = sm[0];
}

// ---------------- row dots -----------------------------------------------------
__global__ void k_rowdot_f32(const float* __restrict__ X, const float* __restrict__ v,
                             float* __restrict__ o, int C) {
  __shared__ float sm[256];
  int r = blockIdx.x;
  float s = 0.f;
  for (int c = threadIdx.x; c < C; c += 256) s += X[(long long)r * C + c] * v[c];
  sm[threadIdx.x] = s; __syncthreads();
  for (int st = 128; st > 0; st >>= 1) {
    if ((int)threadIdx.x < st) sm[threadIdx.x] += sm[threadIdx.x + st];
    __syncthreads();
  }
  if (threadIdx.x == 0) o[r] = sm[0];
}

__global__ void k_rowdot_bf16(const unsigned short* __restrict__ X, int stride, int C,
                              const float* __restrict__ v, float* __restrict__ o) {
  __shared__ float sm[256];
  int r = blockIdx.x;
  float s = 0.f;
  for (int c = threadIdx.x; c < C; c += 256)
    s += bf2f(X[(long long)r * stride + c]) * v[c];
  sm[threadIdx.x] = s; __syncthreads();
  for (int st = 128; st > 0; st >>= 1) {
    if ((int)threadIdx.x < st) sm[threadIdx.x] += sm[threadIdx.x + st];
    __syncthreads();
  }
  if (threadIdx.x == 0) o[r] = sm[0];
}

// ---------------- bf16 WMMA GEMM: C[MxN] = A[MxKp] @ Bt[NxKp]^T ----------------
// block = 128 threads = 4 waves; each wave computes a 16x64 strip (4 accumulators)
// so one A-tile load feeds 4 WMMAs. Block tile = 16 x 256. N % 256 == 0 required.
// A-layout per CDNA5 16-bit A 16x32: lanes0-15 rows M, K {0-7,16-23}; lanes16-31 K {8-15,24-31}.
__global__ void k_gemm(const unsigned short* __restrict__ A,
                       const unsigned short* __restrict__ Bt,
                       const float* __restrict__ bias,
                       float* __restrict__ Cf, unsigned short* __restrict__ Cbf,
                       int M, int N, int Kp, int doRelu) {
  int lane = threadIdx.x & 31;
  int wave = threadIdx.x >> 5;
  int tm = blockIdx.x << 4;
  int tn = (blockIdx.y << 8) + (wave << 6);   // 64-wide strip per wave
  if (tn >= N || tm >= M) return;
  int half = lane >> 4;
  int l15  = lane & 15;
  const size_t bstep = (size_t)16 * Kp;       // one 16-column group of Bt
  const unsigned short* Ap = A  + (size_t)(tm + l15) * Kp + (half << 3);
  const unsigned short* Bp = Bt + (size_t)(tn + l15) * Kp + (half << 3);
  v8f acc0 = {0.f,0.f,0.f,0.f,0.f,0.f,0.f,0.f};
  v8f acc1 = acc0, acc2 = acc0, acc3 = acc0;
  for (int k = 0; k < Kp; k += 32) {
    __builtin_prefetch(Ap + k + 128, 0, 1);
    union { v16bf v; uint4 q[2]; } ua, ub0, ub1, ub2, ub3;
    ua.q[0]  = *(const uint4*)(Ap + k);
    ua.q[1]  = *(const uint4*)(Ap + k + 16);
    ub0.q[0] = *(const uint4*)(Bp + k);
    ub0.q[1] = *(const uint4*)(Bp + k + 16);
    ub1.q[0] = *(const uint4*)(Bp + bstep + k);
    ub1.q[1] = *(const uint4*)(Bp + bstep + k + 16);
    ub2.q[0] = *(const uint4*)(Bp + 2 * bstep + k);
    ub2.q[1] = *(const uint4*)(Bp + 2 * bstep + k + 16);
    ub3.q[0] = *(const uint4*)(Bp + 3 * bstep + k);
    ub3.q[1] = *(const uint4*)(Bp + 3 * bstep + k + 16);
    acc0 = __builtin_amdgcn_wmma_f32_16x16x32_bf16(false, ua.v, false, ub0.v,
                                                   (short)0, acc0, false, false);
    acc1 = __builtin_amdgcn_wmma_f32_16x16x32_bf16(false, ua.v, false, ub1.v,
                                                   (short)0, acc1, false, false);
    acc2 = __builtin_amdgcn_wmma_f32_16x16x32_bf16(false, ua.v, false, ub2.v,
                                                   (short)0, acc2, false, false);
    acc3 = __builtin_amdgcn_wmma_f32_16x16x32_bf16(false, ua.v, false, ub3.v,
                                                   (short)0, acc3, false, false);
  }
  v8f accs[4] = {acc0, acc1, acc2, acc3};
  for (int j = 0; j < 4; ++j) {
    int ocol = tn + (j << 4) + l15;
    float bv = bias ? bias[ocol] : 0.f;
    for (int r = 0; r < 8; ++r) {
      int orow = tm + r + (half << 3);  // C layout: VGPR r -> M=r (lanes0-15), r+8 (16-31)
      float v = accs[j][r] + bv;
      if (doRelu) v = fmaxf(v, 0.f);
      size_t idx = (size_t)orow * N + ocol;
      if (Cf)  Cf[idx]  = v;
      if (Cbf) Cbf[idx] = f2bf(v);
    }
  }
}

// ---------------- fills --------------------------------------------------------
__global__ void k_fill_u32(unsigned* p, unsigned v, long long n) {
  long long i = (long long)blockIdx.x * blockDim.x + threadIdx.x;
  if (i < n) p[i] = v;
}
__global__ void k_fill_f32(float* p, float v, long long n) {
  long long i = (long long)blockIdx.x * blockDim.x + threadIdx.x;
  if (i < n) p[i] = v;
}
__global__ void k_init_bias(float* __restrict__ out, const float* __restrict__ b1,
                            const float* __restrict__ b2, long long rows) {
  long long i = (long long)blockIdx.x * blockDim.x + threadIdx.x;
  if (i >= rows * HD) return;
  int h = (int)(i % HD);
  out[i] = b1[h] + b2[h];
}
__global__ void k_relu(float* p, long long n) {
  long long i = (long long)blockIdx.x * blockDim.x + threadIdx.x;
  if (i < n) p[i] = fmaxf(p[i], 0.f);
}

// ---------------- edge passes --------------------------------------------------
__device__ __forceinline__ void edge_sd(const int* ei, int E, int i, int& s, int& d) {
  if (i < E) { s = ei[i]; d = ei[E + i]; } else { s = d = i - E; }
}

__global__ void k_edge_max(const int* __restrict__ ei, int E, int nloops,
                           const float* __restrict__ ssrc, const float* __restrict__ sdst,
                           unsigned* __restrict__ emax) {
  int i = blockIdx.x * blockDim.x + threadIdx.x;
  if (i >= E + nloops) return;
  int s, d; edge_sd(ei, E, i, s, d);
  float e = ssrc[s] + sdst[d];
  e = (e > 0.f) ? e : 0.2f * e;
  union { float f; int i; } cv; cv.f = e;
  unsigned key = (cv.i >= 0) ? ((unsigned)cv.i | 0x80000000u) : ~(unsigned)cv.i;
  atomicMax(emax + d, key);
}

__global__ void k_fix_emax(float* emax, int n) {
  int i = blockIdx.x * blockDim.x + threadIdx.x;
  if (i >= n) return;
  unsigned key = ((unsigned*)emax)[i];
  unsigned b = (key & 0x80000000u) ? (key ^ 0x80000000u) : ~key;
  union { unsigned u; float f; } cv; cv.u = b;
  float f = cv.f;
  if (!(f >= -1e38f && f <= 1e38f)) f = 0.f;   // non-finite -> 0
  emax[i] = f;
}

__global__ void k_edge_sum(const int* __restrict__ ei, int E, int nloops,
                           const float* __restrict__ ssrc, const float* __restrict__ sdst,
                           const float* __restrict__ emax, float* __restrict__ denom) {
  int i = blockIdx.x * blockDim.x + threadIdx.x;
  if (i >= E + nloops) return;
  int s, d; edge_sd(ei, E, i, s, d);
  float e = ssrc[s] + sdst[d];
  e = (e > 0.f) ? e : 0.2f * e;
  atomicAdd(denom + d, __expf(e - emax[d]));
}

__global__ void k_edge_agg(const int* __restrict__ ei, int E, int nloops,
                           const float* __restrict__ ssrc, const float* __restrict__ sdst,
                           const float* __restrict__ emax, const float* __restrict__ denom,
                           const float* __restrict__ hs, float* __restrict__ out) {
  __shared__ float s_alpha; __shared__ int s_s, s_d;
  int i = blockIdx.x;
  if (threadIdx.x == 0) {
    int s, d; edge_sd(ei, E, i, s, d);
    float e = ssrc[s] + sdst[d];
    e = (e > 0.f) ? e : 0.2f * e;
    float ex = __expf(e - emax[d]);
    s_alpha = ex / (denom[d] + 1e-16f);
    s_s = s; s_d = d;
  }
  __syncthreads();
  float a = s_alpha; int s = s_s, d = s_d;
  const float* hrow = hs + (size_t)s * HD;
  float* orow = out + (size_t)d * HD;
  for (int c = threadIdx.x; c < HD; c += 256)
    atomicAdd(orow + c, a * hrow[c]);
}

// ---------------- MAE cosine loss ---------------------------------------------
__global__ void k_loss(const float* __restrict__ hm, const float* __restrict__ h,
                       const float* __restrict__ mask, float scale, float* acc) {
  __shared__ float sd[256], sa[256], sb[256];
  int r = blockIdx.x;
  float m = mask[r];
  float dd = 0.f, na = 0.f, nb = 0.f;
  for (int c = threadIdx.x; c < HD; c += 256) {
    float a = hm[(size_t)r * HD + c] * m;
    float b = h[(size_t)r * HD + c] * m;
    dd += a * b; na += a * a; nb += b * b;
  }
  sd[threadIdx.x] = dd; sa[threadIdx.x] = na; sb[threadIdx.x] = nb; __syncthreads();
  for (int st = 128; st > 0; st >>= 1) {
    if ((int)threadIdx.x < st) {
      sd[threadIdx.x] += sd[threadIdx.x + st];
      sa[threadIdx.x] += sa[threadIdx.x + st];
      sb[threadIdx.x] += sb[threadIdx.x + st];
    }
    __syncthreads();
  }
  if (threadIdx.x == 0) {
    float cosv = sd[0] / fmaxf(sqrtf(sa[0]) * sqrtf(sb[0]), 1e-8f);
    atomicAdd(acc, (1.f - cosv) * scale);
  }
}

// ---------------- cell l2norm -> bf16 padded ----------------------------------
__global__ void k_cellnorm(const float* __restrict__ X, unsigned short* __restrict__ out) {
  __shared__ float sm[256]; __shared__ float s_inv;
  int r = blockIdx.x;
  float s = 0.f;
  for (int c = threadIdx.x; c < FC; c += 256) { float v = X[(size_t)r * FC + c]; s += v * v; }
  sm[threadIdx.x] = s; __syncthreads();
  for (int st = 128; st > 0; st >>= 1) {
    if ((int)threadIdx.x < st) sm[threadIdx.x] += sm[threadIdx.x + st];
    __syncthreads();
  }
  if (threadIdx.x == 0) s_inv = 1.f / fmaxf(sqrtf(sm[0]), 1e-12f);
  __syncthreads();
  for (int c = threadIdx.x; c < FCP; c += 256) {
    float v = (c < FC) ? X[(size_t)r * FC + c] * s_inv : 0.f;
    out[(size_t)r * FCP + c] = f2bf(v);
  }
}

// ---------------- concat [hd[d1], hd[d2], cell] l2norm -> bf16 ----------------
__global__ void k_hcat(const float* __restrict__ hd, const float* __restrict__ c3,
                       const int* __restrict__ id1, const int* __restrict__ id2,
                       unsigned short* __restrict__ out) {
  __shared__ float sm[256]; __shared__ float s_inv;
  int r = blockIdx.x;
  int d1 = id1[r], d2 = id2[r];
  float s = 0.f;
  for (int c = threadIdx.x; c < 3072; c += 256) {
    float v = (c < 768) ? hd[(size_t)d1 * HD + c]
            : (c < 1536) ? hd[(size_t)d2 * HD + (c - 768)]
            : c3[(size_t)r * 1536 + (c - 1536)];
    s += v * v;
  }
  sm[threadIdx.x] = s; __syncthreads();
  for (int st = 128; st > 0; st >>= 1) {
    if ((int)threadIdx.x < st) sm[threadIdx.x] += sm[threadIdx.x + st];
    __syncthreads();
  }
  if (threadIdx.x == 0) s_inv = 1.f / fmaxf(sqrtf(sm[0]), 1e-12f);
  __syncthreads();
  for (int c = threadIdx.x; c < 3072; c += 256) {
    float v = (c < 768) ? hd[(size_t)d1 * HD + c]
            : (c < 1536) ? hd[(size_t)d2 * HD + (c - 768)]
            : c3[(size_t)r * 1536 + (c - 1536)];
    out[(size_t)r * 3072 + c] = f2bf(v * s_inv);
  }
}

// ---------------- final 768 -> 2 head ------------------------------------------
__global__ void k_head(const float* __restrict__ h3, const float* __restrict__ cW,
                       const float* __restrict__ cb, float* __restrict__ out, int rows) {
  int r = blockIdx.x * blockDim.x + threadIdx.x;
  if (r >= rows) return;
  float o0 = cb[0], o1 = cb[1];
  for (int h = 0; h < HD; ++h) {
    float v = h3[(size_t)r * HD + h];
    o0 += v * cW[h * 2];
    o1 += v * cW[h * 2 + 1];
  }
  out[r * 2] = o0; out[r * 2 + 1] = o1;
}

__global__ void k_copy1(const float* a, float* dst) {
  if (blockIdx.x == 0 && threadIdx.x == 0) dst[0] = a[0];
}

// ==============================================================================
extern "C" void kernel_launch(void* const* d_in, const int* in_sizes, int n_in,
                              void* d_out, int out_size, void* d_ws, size_t ws_size,
                              hipStream_t stream) {
  const int*   d1id  = (const int*)d_in[0];
  const int*   d2id  = (const int*)d_in[1];
  const float* cellf = (const float*)d_in[2];
  const float* xd    = (const float*)d_in[3];
  const float* xt    = (const float*)d_in[4];
  const int*   ei_dd = (const int*)d_in[5];
  const int*   ei_dt = (const int*)d_in[6];
  const int*   ei_td = (const int*)d_in[7];
  const int*   ei_tt = (const int*)d_in[8];
  const float* dmask = (const float*)d_in[9];
  const float* tmask = (const float*)d_in[10];
  #define P(i) ((const float*)d_in[(i)])
  float* dout = (float*)d_out;

  // ---- workspace bump allocator ----
  size_t off = 0;
  auto alloc = [&](size_t bytes) -> void* {
    void* p = (char*)d_ws + off;
    off += (bytes + 255) & ~(size_t)255;
    return p;
  };
  unsigned short* xd_bf  = (unsigned short*)alloc((size_t)ND * FDP * 2);
  unsigned short* xdm_bf = (unsigned short*)alloc((size_t)ND * FDP * 2);
  unsigned short* xt_bf  = (unsigned short*)alloc((size_t)NT * FTP * 2);
  unsigned short* xtm_bf = (unsigned short*)alloc((size_t)NT * FTP * 2);
  unsigned short* w_dd   = (unsigned short*)alloc((size_t)HD * FDP * 2);
  unsigned short* w_dt   = (unsigned short*)alloc((size_t)HD * FDP * 2);
  unsigned short* w_td   = (unsigned short*)alloc((size_t)HD * FTP * 2);
  unsigned short* w_tt   = (unsigned short*)alloc((size_t)HD * FTP * 2);
  float* wdv_dd = (float*)alloc(FDP * 4);
  float* wdv_dt = (float*)alloc(FTP * 4);
  float* wdv_td = (float*)alloc(FDP * 4);
  float* wdv_tt = (float*)alloc(FTP * 4);
  float* hs_d  = (float*)alloc((size_t)ND * HD * 4);
  float* hs_t  = (float*)alloc((size_t)NT * HD * 4);
  float* ssrc  = (float*)alloc((size_t)NT * 4);
  float* sdst  = (float*)alloc((size_t)NT * 4);
  float* emax  = (float*)alloc((size_t)NT * 4);
  float* denom = (float*)alloc((size_t)NT * 4);
  float* hd_m  = (float*)alloc((size_t)ND * HD * 4);
  float* ht_m  = (float*)alloc((size_t)NT * HD * 4);
  float* hdc   = (float*)alloc((size_t)ND * HD * 4);
  float* htc   = (float*)alloc((size_t)NT * HD * 4);
  float* lacc  = (float*)alloc(256);
  unsigned short* celln = (unsigned short*)alloc((size_t)BSZ * FCP * 2);
  unsigned short* rW1t  = (unsigned short*)alloc((size_t)2048 * FCP * 2);
  unsigned short* rW2t  = (unsigned short*)alloc((size_t)512 * 2048 * 2);
  unsigned short* rW3t  = (unsigned short*)alloc((size_t)1536 * 512 * 2);
  unsigned short* sW1t  = (unsigned short*)alloc((size_t)2048 * 3072 * 2);
  unsigned short* sW2t  = (unsigned short*)alloc((size_t)512 * 2048 * 2);
  unsigned short* sW3t  = (unsigned short*)alloc((size_t)768 * 512 * 2);
  unsigned short* a2048 = (unsigned short*)alloc((size_t)BSZ * 2048 * 2);
  unsigned short* a512  = (unsigned short*)alloc((size_t)BSZ * 512 * 2);
  float* c3f  = (float*)alloc((size_t)BSZ * 1536 * 4);
  unsigned short* hcat = (unsigned short*)alloc((size_t)BSZ * 3072 * 2);
  float* h3f  = (float*)alloc((size_t)BSZ * HD * 4);

  auto gblk = [](long long n) { return (unsigned)((n + 255) / 256); };

  // ---- 1. input prep (clean + masked bf16, padded) ----
  k_prep<<<gblk((long long)ND * FDP), 256, 0, stream>>>(xd, dmask, P(31), xd_bf, xdm_bf, ND, FD, FDP);
  k_prep<<<gblk((long long)NT * FTP), 256, 0, stream>>>(xt, tmask, P(32), xt_bf, xtm_bf, NT, FT, FTP);

  // ---- 2. weight conversions (transposed N x Kp bf16) ----
  k_wconv_t<<<gblk((long long)HD * FDP), 256, 0, stream>>>(P(11), w_dd, FD, HD, FDP);
  k_wconv_t<<<gblk((long long)HD * FDP), 256, 0, stream>>>(P(16), w_dt, FD, HD, FDP);
  k_wconv_t<<<gblk((long long)HD * FTP), 256, 0, stream>>>(P(21), w_td, FT, HD, FTP);
  k_wconv_t<<<gblk((long long)HD * FTP), 256, 0, stream>>>(P(26), w_tt, FT, HD, FTP);
  k_wconv_t<<<gblk((long long)2048 * FCP), 256, 0, stream>>>(P(33), rW1t, FC, 2048, FCP);
  k_wconv_t<<<gblk((long long)512 * 2048), 256, 0, stream>>>(P(35), rW2t, 2048, 512, 2048);
  k_wconv_t<<<gblk((long long)1536 * 512), 256, 0, stream>>>(P(37), rW3t, 512, 1536, 512);
  k_wconv_t<<<gblk((long long)2048 * 3072), 256, 0, stream>>>(P(39), sW1t, 3072, 2048, 3072);
  k_wconv_t<<<gblk((long long)512 * 2048), 256, 0, stream>>>(P(41), sW2t, 2048, 512, 2048);
  k_wconv_t<<<gblk((long long)768 * 512), 256, 0, stream>>>(P(43), sW3t, 512, 768, 512);

  // ---- 3. folded dst projections: wdv = Wdst @ adst ----
  k_colvec<<<FD, 256, 0, stream>>>(P(12), P(14), wdv_dd, HD);
  k_colvec<<<FT, 256, 0, stream>>>(P(17), P(19), wdv_dt, HD);
  k_colvec<<<FD, 256, 0, stream>>>(P(22), P(24), wdv_td, HD);
  k_colvec<<<FT, 256, 0, stream>>>(P(27), P(29), wdv_tt, HD);

  const unsigned NEG_INF_KEY = 0x007FFFFFu;  // encoded -inf for atomic float-max

  auto relation = [&](const unsigned short* src_bf, int src_rows, int src_Kp,
                      const unsigned short* dst_bf, int dst_rows, int dst_Kp, int dst_K,
                      const unsigned short* Wsrc_t, const float* asrc, const float* wdv,
                      const int* ei, int E, int nloops, float* hsbuf, float* outbuf) {
    // hs = src @ Wsrc  (bf16 WMMA)  -- block tile 16x256 (4 waves x 16x64)
    dim3 gg(src_rows / 16, HD / 256);
    k_gemm<<<gg, 128, 0, stream>>>(src_bf, Wsrc_t, nullptr, hsbuf, nullptr,
                                   src_rows, HD, src_Kp, 0);
    // ssrc = hs . asrc ; sdst = x_dst . (Wdst @ adst)
    k_rowdot_f32<<<src_rows, 256, 0, stream>>>(hsbuf, asrc, ssrc, HD);
    k_rowdot_bf16<<<dst_rows, 256, 0, stream>>>(dst_bf, dst_Kp, dst_K, wdv, sdst);
    // softmax over incoming edges
    int total = E + nloops;
    k_fill_u32<<<gblk(dst_rows), 256, 0, stream>>>((unsigned*)emax, NEG_INF_KEY, dst_rows);
    k_edge_max<<<gblk(total), 256, 0, stream>>>(ei, E, nloops, ssrc, sdst, (unsigned*)emax);
    k_fix_emax<<<gblk(dst_rows), 256, 0, stream>>>(emax, dst_rows);
    k_fill_f32<<<gblk(dst_rows), 256, 0, stream>>>(denom, 0.f, dst_rows);
    k_edge_sum<<<gblk(total), 256, 0, stream>>>(ei, E, nloops, ssrc, sdst, emax, denom);
    k_edge_agg<<<(unsigned)total, 256, 0, stream>>>(ei, E, nloops, ssrc, sdst, emax, denom,
                                                    hsbuf, outbuf);
  };

  auto branch = [&](const unsigned short* xdb, const unsigned short* xtb,
                    float* outD, float* outT) {
    k_init_bias<<<gblk((long long)ND * HD), 256, 0, stream>>>(outD, P(15), P(25), ND);
    k_init_bias<<<gblk((long long)NT * HD), 256, 0, stream>>>(outT, P(20), P(30), NT);
    relation(xdb, ND, FDP, xdb, ND, FDP, FD, w_dd, P(13), wdv_dd, ei_dd, N_EDD, ND, hs_d, outD);
    relation(xdb, ND, FDP, xtb, NT, FTP, FT, w_dt, P(18), wdv_dt, ei_dt, N_EDT, 0,  hs_d, outT);
    relation(xtb, NT, FTP, xdb, ND, FDP, FD, w_td, P(23), wdv_td, ei_td, N_ETD, 0,  hs_t, outD);
    relation(xtb, NT, FTP, xtb, NT, FTP, FT, w_tt, P(28), wdv_tt, ei_tt, N_ETT, NT, hs_t, outT);
    k_relu<<<gblk((long long)ND * HD), 256, 0, stream>>>(outD, (long long)ND * HD);
    k_relu<<<gblk((long long)NT * HD), 256, 0, stream>>>(outT, (long long)NT * HD);
  };

  // ---- 4. masked + clean GNN branches ----
  branch(xdm_bf, xtm_bf, hd_m, ht_m);
  branch(xd_bf,  xt_bf,  hdc,  htc);

  // ---- 5. MAE loss ----
  k_fill_f32<<<1, 256, 0, stream>>>(lacc, 0.f, 1);
  k_loss<<<ND, 256, 0, stream>>>(hd_m, hdc, dmask, 1.f / (float)ND, lacc);
  k_loss<<<NT, 256, 0, stream>>>(ht_m, htc, tmask, 1.f / (float)NT, lacc);
  k_copy1<<<1, 1, 0, stream>>>(lacc, dout + BSZ * 2);

  // ---- 6. cell MLP (890->2048->512->1536) ----
  k_cellnorm<<<BSZ, 256, 0, stream>>>(cellf, celln);
  k_gemm<<<dim3(BSZ / 16, 2048 / 256), 128, 0, stream>>>(celln, rW1t, P(34), nullptr, a2048,
                                                         BSZ, 2048, FCP, 1);
  k_gemm<<<dim3(BSZ / 16, 512 / 256), 128, 0, stream>>>(a2048, rW2t, P(36), nullptr, a512,
                                                        BSZ, 512, 2048, 1);
  k_gemm<<<dim3(BSZ / 16, 1536 / 256), 128, 0, stream>>>(a512, rW3t, P(38), c3f, nullptr,
                                                         BSZ, 1536, 512, 1);

  // ---- 7. concat + l2norm, hidden MLP (3072->2048->512->768) ----
  k_hcat<<<BSZ, 256, 0, stream>>>(hdc, c3f, d1id, d2id, hcat);
  k_gemm<<<dim3(BSZ / 16, 2048 / 256), 128, 0, stream>>>(hcat, sW1t, P(40), nullptr, a2048,
                                                         BSZ, 2048, 3072, 1);
  k_gemm<<<dim3(BSZ / 16, 512 / 256), 128, 0, stream>>>(a2048, sW2t, P(42), nullptr, a512,
                                                        BSZ, 512, 2048, 1);
  k_gemm<<<dim3(BSZ / 16, 768 / 256), 128, 0, stream>>>(a512, sW3t, P(44), h3f, nullptr,
                                                        BSZ, 768, 512, 1);

  // ---- 8. final head 768 -> 2 ----
  k_head<<<gblk(BSZ), 256, 0, stream>>>(h3f, P(45), P(46), dout, BSZ);
  #undef P
}
